// PyramidAttention_30958124269972
// MI455X (gfx1250) — matrix operand order repaired and checked by reading, not compiled, not run on hardware
//
#include <hip/hip_runtime.h>
#include <hip/hip_bf16.h>
#include <math.h>

// ---------------- types / constants ----------------
typedef _Float16 v16h __attribute__((ext_vector_type(16)));
typedef _Float16 v8h  __attribute__((ext_vector_type(8)));
typedef float    v8f  __attribute__((ext_vector_type(8)));

#define BIC_A (-0.75f)
static constexpr int Bn = 2, Cc = 64, CmC = 32, HW = 4096;
static constexpr int Ltot = 13326;
static constexpr int LP   = 13344;            // padded to mult of 32
static constexpr int NT16 = LP / 16;          // 834 key tiles (pass A), split 417/417
static constexpr int NT32 = LP / 32;          // 417 key tiles (pass B)

__device__ __forceinline__ float cubw(float d) {
    d = fabsf(d);
    if (d <= 1.0f) return ((BIC_A + 2.0f) * d - (BIC_A + 3.0f)) * d * d + 1.0f;
    if (d <  2.0f) return ((BIC_A * d - 5.0f * BIC_A) * d + 8.0f * BIC_A) * d - 4.0f * BIC_A;
    return 0.0f;
}

__device__ __forceinline__ v16h ld_frag(const _Float16* p0, const _Float16* p1) {
    union { v16h v; v8h h[2]; } u;
    u.h[0] = *reinterpret_cast<const v8h*>(p0);
    u.h[1] = *reinterpret_cast<const v8h*>(p1);
    return u.v;
}

// Force all 9 (resp. 6) fragments simultaneously live so the loads issue as one
// clause (single s_wait) and the WMMAs run back-to-back.
__device__ __forceinline__ void force_live9(v16h* a) {
    asm volatile("" : "+v"(a[0]), "+v"(a[1]), "+v"(a[2]), "+v"(a[3]), "+v"(a[4]),
                      "+v"(a[5]), "+v"(a[6]), "+v"(a[7]), "+v"(a[8]));
}
__device__ __forceinline__ void force_live6(v16h* a) {
    asm volatile("" : "+v"(a[0]), "+v"(a[1]), "+v"(a[2]), "+v"(a[3]), "+v"(a[4]),
                      "+v"(a[5]));
}

// ---------------- stage 1: bicubic resize (align_corners) ----------------
__global__ void k_resize(const float* __restrict__ src, float* __restrict__ dst,
                         int Hs, int Ns, int total) {
    int t = blockIdx.x * blockDim.x + threadIdx.x;
    if (t >= total) return;
    int pix = t % Ns, bc = t / Ns;
    int oy = pix / Hs, ox = pix % Hs;
    float sc = 63.0f / (float)(Hs - 1);
    float ry = oy * sc, rx = ox * sc;
    int iy0 = (int)floorf(ry), ix0 = (int)floorf(rx);
    float fy = ry - iy0, fx = rx - ix0;
    float wy[4] = { cubw(fy + 1.0f), cubw(fy), cubw(1.0f - fy), cubw(2.0f - fy) };
    float wx[4] = { cubw(fx + 1.0f), cubw(fx), cubw(1.0f - fx), cubw(2.0f - fx) };
    const float* p = src + (size_t)bc * HW;
    float acc = 0.0f;
    #pragma unroll
    for (int i = 0; i < 4; ++i) {
        int yy = min(max(iy0 - 1 + i, 0), 63);
        #pragma unroll
        for (int j = 0; j < 4; ++j) {
            int xx = min(max(ix0 - 1 + j, 0), 63);
            acc += wy[i] * wx[j] * p[yy * 64 + xx];
        }
    }
    dst[(size_t)bc * Ns + pix] = acc;
}

// ---------------- stage 2: 1x1 conv + PReLU -> f16 ----------------
__global__ void k_conv1x1(const float* __restrict__ in, const float* __restrict__ Wt,
                          const float* __restrict__ bias, const float* __restrict__ ap,
                          _Float16* __restrict__ out, int Cout, int N, int total) {
    int t = blockIdx.x * blockDim.x + threadIdx.x;
    if (t >= total) return;
    int pix = t % N, o = (t / N) % Cout, b = t / (N * Cout);
    const float* ip = in + (size_t)b * Cc * N + pix;
    const float* wr = Wt + o * Cc;
    float acc = bias[o];
    #pragma unroll 8
    for (int c = 0; c < Cc; ++c) acc += wr[c] * ip[(size_t)c * N];
    float a = ap[0];
    acc = (acc >= 0.0f) ? acc : a * acc;
    out[((size_t)b * Cout + o) * N + pix] = (_Float16)acc;
}

// ---------------- stage 3: patch (im2col) matrices ----------------
__global__ void k_patch_rm(const _Float16* __restrict__ src, _Float16* __restrict__ dst,
                           int Cch, int Hs, int Ns, int Kp, size_t rowsTot, int rofs, int total) {
    int t = blockIdx.x * blockDim.x + threadIdx.x;
    if (t >= total) return;
    int k = t % Kp, ll = (t / Kp) % Ns, b = t / (Kp * Ns);
    int c = k / 9, j = k % 9, dy = j / 3 - 1, dx = j % 3 - 1;
    int y = ll / Hs, x = ll % Hs, ny = y + dy, nx = x + dx;
    _Float16 v = (_Float16)0.0f;
    if ((unsigned)ny < (unsigned)Hs && (unsigned)nx < (unsigned)Hs)
        v = src[((size_t)b * Cch + c) * Ns + ny * Hs + nx];
    dst[((size_t)b * rowsTot + rofs + ll) * Kp + k] = v;
}

// K-major (transposed): dst[(b*576 + k)*LP + lofs + ll]   (base patches, C=64)
__global__ void k_patch_cm(const _Float16* __restrict__ src, _Float16* __restrict__ dst,
                           int Hs, int Ns, int lofs, int total) {
    int t = blockIdx.x * blockDim.x + threadIdx.x;
    if (t >= total) return;
    int k = t % 576, ll = (t / 576) % Ns, b = t / (576 * Ns);
    int c = k / 9, j = k % 9, dy = j / 3 - 1, dx = j % 3 - 1;
    int y = ll / Hs, x = ll % Hs, ny = y + dy, nx = x + dx;
    _Float16 v = (_Float16)0.0f;
    if ((unsigned)ny < (unsigned)Hs && (unsigned)nx < (unsigned)Hs)
        v = src[((size_t)b * Cc + c) * Ns + ny * Hs + nx];
    dst[((size_t)b * 576 + k) * LP + lofs + ll] = v;
}

__global__ void k_zero_rp(_Float16* __restrict__ RpT, int total) {  // pad cols L..LP
    int t = blockIdx.x * blockDim.x + threadIdx.x;
    if (t >= total) return;
    int pad = LP - Ltot;
    int i = t % pad, k = (t / pad) % 576, b = t / (pad * 576);
    RpT[((size_t)b * 576 + k) * LP + Ltot + i] = (_Float16)0.0f;
}

// ---------------- stage 4: key norms (fold 1/||w|| and softmax scale 10) ----------------
__global__ void k_norm(const _Float16* __restrict__ Wpat, float* __restrict__ inv, int total) {
    int t = blockIdx.x * blockDim.x + threadIdx.x;
    if (t >= total) return;
    int l = t % Ltot, b = t / Ltot;
    const _Float16* r = Wpat + ((size_t)b * LP + l) * 288;
    float s = 0.0f;
    #pragma unroll 8
    for (int k = 0; k < 288; ++k) { float v = (float)r[k]; s += v * v; }
    float n = fmaxf(sqrtf(s), 1e-4f);
    inv[(size_t)b * LP + l] = 10.0f / n;
}

// ---------------- stage 5 (pass A): streaming score GEMM + online max/sumexp ----------------
// 2 waves per 16-query tile; each wave scans half of the key range.
// Per-LANE online softmax (each lane's key column is a fixed subset) — no
// cross-lane traffic in the hot loop; single log2 merge at the end.
__global__ __launch_bounds__(64) void k_scores_stats(
        const _Float16* __restrict__ Apat, const _Float16* __restrict__ Wpat,
        const float* __restrict__ inv, float* __restrict__ rowM, float* __restrict__ rowZ) {
    __shared__ float sm[2][16], sz[2][16];
    int b = blockIdx.x >> 8;
    int q0 = (blockIdx.x & 255) << 4;
    int w = threadIdx.x >> 5, lane = threadIdx.x & 31;
    int mrow = lane & 15, hi = lane >> 4;

    // A fragments: 16 queries x 288, resident in VGPRs for the whole key loop
    v16h afr[9];
    const _Float16* abase = Apat + ((size_t)b * HW + q0 + mrow) * 288;
    #pragma unroll
    for (int ks = 0; ks < 9; ++ks) {
        const _Float16* p = abase + ks * 32 + hi * 8;
        afr[ks] = ld_frag(p, p + 16);
    }

    float rm[8], rs[8];
    #pragma unroll
    for (int r = 0; r < 8; ++r) { rm[r] = -1e30f; rs[r] = 0.0f; }

    int lt0 = w * (NT16 / 2), lt1 = lt0 + (NT16 / 2);
    for (int lt = lt0; lt < lt1; ++lt) {
        int myl = lt * 16 + mrow;
        float iv = (myl < Ltot) ? inv[(size_t)b * LP + myl] : 0.0f;
        const _Float16* bbase = Wpat + ((size_t)b * LP + myl) * 288 + hi * 8;
        v16h bfr[9];
        #pragma unroll
        for (int ks = 0; ks < 9; ++ks)
            bfr[ks] = ld_frag(bbase + ks * 32, bbase + ks * 32 + 16);
        force_live9(bfr);                 // one load clause, one wait, 9 WMMAs
        v8f c = {};
        #pragma unroll
        for (int ks = 0; ks < 9; ++ks)
            c = __builtin_amdgcn_wmma_f32_16x16x32_f16(false, afr[ks], false, bfr[ks],
                                                       (short)0, c, false, false);
        #pragma unroll
        for (int r = 0; r < 8; ++r) {
            float s = (myl < Ltot) ? c[r] * iv : -1e30f;
            float mn = fmaxf(rm[r], s);
            rs[r] = rs[r] * expf(rm[r] - mn) + expf(s - mn);
            rm[r] = mn;
        }
    }
    // merge (m,z) across the 16-lane group (disjoint key subsets per lane)
    #pragma unroll
    for (int r = 0; r < 8; ++r) {
        float m = rm[r], z = rs[r];
        #pragma unroll
        for (int d = 1; d < 16; d <<= 1) {
            float mo = __shfl_xor(m, d);
            float zo = __shfl_xor(z, d);
            float mn = fmaxf(m, mo);
            z = z * expf(m - mn) + zo * expf(mo - mn);
            m = mn;
        }
        rm[r] = m; rs[r] = z;
    }
    if (mrow == 0) {
        #pragma unroll
        for (int r = 0; r < 8; ++r) { sm[w][8 * hi + r] = rm[r]; sz[w][8 * hi + r] = rs[r]; }
    }
    __syncthreads();
    if (threadIdx.x < 16) {
        int i = threadIdx.x;
        float m1 = sm[0][i], m2 = sm[1][i];
        float m = fmaxf(m1, m2);
        float z = sz[0][i] * expf(m1 - m) + sz[1][i] * expf(m2 - m);
        rowM[(size_t)b * HW + q0 + i] = m;
        rowZ[(size_t)b * HW + q0 + i] = z;
    }
}

// ---------------- stage 6 (pass B): rescore -> softmax -> output WMMA GEMM ----------------
// 8 waves / block, double-buffered Y tile in LDS.
//   waves 0,1: producers — rescore one 16-key half each, softmax, write Y (f16)
//   waves 2..7: consumers — 6 of the 36 (shift, channel-block) accumulators each
__global__ __launch_bounds__(256) void k_attn_out(
        const _Float16* __restrict__ Apat, const _Float16* __restrict__ Wpat,
        const _Float16* __restrict__ RpT,  const float* __restrict__ inv,
        const float* __restrict__ rowM,    const float* __restrict__ rowZ,
        float* __restrict__ tbuf) {
    __shared__ __align__(16) _Float16 Yt[2][16 * 32];
    int b  = blockIdx.x >> 8;
    int q0 = (blockIdx.x & 255) << 4;
    int lane = threadIdx.x & 31, w = threadIdx.x >> 5;
    int mrow = lane & 15, hi = lane >> 4;

    v16h afr[9];
    float m8[8], rz8[8];
    if (w < 2) {
        const _Float16* abase = Apat + ((size_t)b * HW + q0 + mrow) * 288;
        #pragma unroll
        for (int ks = 0; ks < 9; ++ks) {
            const _Float16* p = abase + ks * 32 + hi * 8;
            afr[ks] = ld_frag(p, p + 16);
        }
        #pragma unroll
        for (int r = 0; r < 8; ++r) {
            int q = q0 + 8 * hi + r;
            m8[r]  = rowM[(size_t)b * HW + q];
            rz8[r] = 1.0f / fmaxf(rowZ[(size_t)b * HW + q], 1e-30f);
        }
    }

    // producer: score+softmax 16 keys (half `w` of tile kt) into Yt[buf]
    auto produce = [&](int kt, int buf) {
        int myl = kt * 32 + w * 16 + mrow;
        float iv = (myl < Ltot) ? inv[(size_t)b * LP + myl] : 0.0f;
        const _Float16* bbase = Wpat + ((size_t)b * LP + myl) * 288 + hi * 8;
        v16h bfr[9];
        #pragma unroll
        for (int ks = 0; ks < 9; ++ks)
            bfr[ks] = ld_frag(bbase + ks * 32, bbase + ks * 32 + 16);
        force_live9(bfr);
        v8f c = {};
        #pragma unroll
        for (int ks = 0; ks < 9; ++ks)
            c = __builtin_amdgcn_wmma_f32_16x16x32_f16(false, afr[ks], false, bfr[ks],
                                                       (short)0, c, false, false);
        #pragma unroll
        for (int r = 0; r < 8; ++r) {
            float y = (myl < Ltot) ? expf(c[r] * iv - m8[r]) * rz8[r] : 0.0f;
            Yt[buf][(r + 8 * hi) * 32 + w * 16 + mrow] = (_Float16)y;
        }
    };

    v8f acc[6];
    #pragma unroll
    for (int i = 0; i < 6; ++i) acc[i] = (v8f){};

    if (w < 2) produce(0, 0);
    __syncthreads();

    for (int kt = 0; kt < NT32; ++kt) {
        int cur = kt & 1;
        if (w >= 2) {
            int l0 = kt * 32;
            const _Float16* yp = &Yt[cur][mrow * 32 + hi * 8];
            v16h a2 = ld_frag(yp, yp + 16);
            v16h bfr[6];
            #pragma unroll
            for (int i = 0; i < 6; ++i) {
                int id = (w - 2) * 6 + i;
                int s = id >> 2, cb = id & 3;
                int k576 = (cb * 16 + mrow) * 9 + (8 - s);     // flipped kernel tap
                const _Float16* bp = RpT + ((size_t)b * 576 + k576) * LP + l0 + hi * 8;
                bfr[i] = ld_frag(bp, bp + 16);
            }
            force_live6(bfr);
            #pragma unroll
            for (int i = 0; i < 6; ++i)
                acc[i] = __builtin_amdgcn_wmma_f32_16x16x32_f16(
                    false, a2, false, bfr[i], (short)0, acc[i], false, false);
        } else if (kt + 1 < NT32) {
            produce(kt + 1, cur ^ 1);
        }
        __syncthreads();
    }

    if (w >= 2) {
        #pragma unroll
        for (int i = 0; i < 6; ++i) {
            int id = (w - 2) * 6 + i;
            int s = id >> 2, cb = id & 3;
            #pragma unroll
            for (int r = 0; r < 8; ++r) {
                int q = q0 + 8 * hi + r;
                tbuf[(((size_t)b * HW + q) * 9 + s) * 64 + cb * 16 + mrow] = acc[i][r];
            }
        }
    }
}

// ---------------- stage 7: 9-tap gather, /4, + residual ----------------
__global__ void k_final(const float* __restrict__ in, const float* __restrict__ tbuf,
                        float* __restrict__ out, int total) {
    int t = blockIdx.x * blockDim.x + threadIdx.x;
    if (t >= total) return;
    int x = t & 63, y = (t >> 6) & 63, c = (t >> 12) & 63, b = t >> 18;
    float acc = 0.0f;
    #pragma unroll
    for (int u = 0; u < 3; ++u) {
        int ny = y + u - 1;
        if ((unsigned)ny >= 64u) continue;
        #pragma unroll
        for (int v = 0; v < 3; ++v) {
            int nx = x + v - 1;
            if ((unsigned)nx >= 64u) continue;
            acc += tbuf[(((size_t)b * HW + ny * 64 + nx) * 9 + (u * 3 + v)) * 64 + c];
        }
    }
    out[t] = in[t] + 0.25f * acc;
}

// ---------------- host ----------------
extern "C" void kernel_launch(void* const* d_in, const int* in_sizes, int n_in,
                              void* d_out, int out_size, void* d_ws, size_t ws_size,
                              hipStream_t stream) {
    (void)in_sizes; (void)n_in; (void)out_size; (void)ws_size;
    const float* input = (const float*)d_in[0];
    const float* inref = (const float*)d_in[1];
    const float* Wb = (const float*)d_in[2];
    const float* bb = (const float*)d_in[3];
    const float* Wm = (const float*)d_in[4];
    const float* bm = (const float*)d_in[5];
    const float* Wa = (const float*)d_in[6];
    const float* ba = (const float*)d_in[7];
    const float* ap = (const float*)d_in[8];
    float* out = (float*)d_out;
    char* ws = (char*)d_ws;

    static const int HSs[5]  = {64, 57, 51, 44, 38};
    static const int NSs[5]  = {4096, 3249, 2601, 1936, 1444};
    static const int LOFS[5] = {0, 4096, 7345, 9946, 11882};

    size_t off = 0;
    auto alloc = [&](size_t bytes) { size_t o = off; off = (off + bytes + 255) & ~(size_t)255; return o; };

    size_t o_ref[5]; o_ref[0] = 0;
    for (int s = 1; s < 5; ++s) o_ref[s] = alloc((size_t)Bn * Cc * NSs[s] * 4);
    size_t o_mb = alloc((size_t)Bn * CmC * HW * 2);
    size_t o_m[5], o_ba[5];
    for (int s = 0; s < 5; ++s) {
        o_m[s]  = alloc((size_t)Bn * CmC * NSs[s] * 2);
        o_ba[s] = alloc((size_t)Bn * Cc  * NSs[s] * 2);
    }
    size_t o_Ap  = alloc((size_t)Bn * HW * 288 * 2);
    size_t o_Wp  = alloc((size_t)Bn * LP * 288 * 2);
    size_t o_Rp  = alloc((size_t)Bn * 576 * LP * 2);
    size_t o_inv = alloc((size_t)Bn * LP * 4);
    size_t o_rm  = alloc((size_t)Bn * HW * 4);
    size_t o_rz  = alloc((size_t)Bn * HW * 4);
    size_t o_t   = alloc((size_t)Bn * HW * 9 * 64 * 4);

    const int TH = 256;
    auto G = [&](int n) { return dim3((unsigned)((n + TH - 1) / TH)); };

    // 1) bicubic pyramid
    for (int s = 1; s < 5; ++s) {
        int tot = Bn * Cc * NSs[s];
        k_resize<<<G(tot), TH, 0, stream>>>(inref, (float*)(ws + o_ref[s]), HSs[s], NSs[s], tot);
    }
    // 2) 1x1 convs + PReLU
    k_conv1x1<<<G(Bn * CmC * HW), TH, 0, stream>>>(input, Wb, bb, ap,
        (_Float16*)(ws + o_mb), CmC, HW, Bn * CmC * HW);
    for (int s = 0; s < 5; ++s) {
        const float* ref = s ? (const float*)(ws + o_ref[s]) : inref;
        k_conv1x1<<<G(Bn * CmC * NSs[s]), TH, 0, stream>>>(ref, Wm, bm, ap,
            (_Float16*)(ws + o_m[s]), CmC, NSs[s], Bn * CmC * NSs[s]);
        k_conv1x1<<<G(Bn * Cc * NSs[s]), TH, 0, stream>>>(ref, Wa, ba, ap,
            (_Float16*)(ws + o_ba[s]), Cc, NSs[s], Bn * Cc * NSs[s]);
    }
    // 3) patch matrices (zero pad rows/cols first)
    for (int b = 0; b < Bn; ++b)
        hipMemsetAsync(ws + o_Wp + ((size_t)b * LP + Ltot) * 288 * 2, 0,
                       (size_t)(LP - Ltot) * 288 * 2, stream);
    {
        int tot = Bn * 576 * (LP - Ltot);
        k_zero_rp<<<G(tot), TH, 0, stream>>>((_Float16*)(ws + o_Rp), tot);
    }
    k_patch_rm<<<G(Bn * HW * 288), TH, 0, stream>>>((const _Float16*)(ws + o_mb),
        (_Float16*)(ws + o_Ap), CmC, 64, HW, 288, (size_t)HW, 0, Bn * HW * 288);
    for (int s = 0; s < 5; ++s) {
        int tot1 = Bn * NSs[s] * 288;
        k_patch_rm<<<G(tot1), TH, 0, stream>>>((const _Float16*)(ws + o_m[s]),
            (_Float16*)(ws + o_Wp), CmC, HSs[s], NSs[s], 288, (size_t)LP, LOFS[s], tot1);
        int tot2 = Bn * NSs[s] * 576;
        k_patch_cm<<<G(tot2), TH, 0, stream>>>((const _Float16*)(ws + o_ba[s]),
            (_Float16*)(ws + o_Rp), HSs[s], NSs[s], LOFS[s], tot2);
    }
    // 4) key norms
    k_norm<<<G(Bn * Ltot), TH, 0, stream>>>((const _Float16*)(ws + o_Wp),
        (float*)(ws + o_inv), Bn * Ltot);
    // 5) pass A: streaming WMMA scores + online softmax stats (2 waves / q-tile)
    k_scores_stats<<<dim3(Bn * 256), dim3(64), 0, stream>>>(
        (const _Float16*)(ws + o_Ap), (const _Float16*)(ws + o_Wp),
        (const float*)(ws + o_inv), (float*)(ws + o_rm), (float*)(ws + o_rz));
    // 6) pass B: rescore + softmax + output WMMA GEMM (2 producer + 6 consumer waves)
    k_attn_out<<<dim3(Bn * 256), dim3(256), 0, stream>>>(
        (const _Float16*)(ws + o_Ap), (const _Float16*)(ws + o_Wp),
        (const _Float16*)(ws + o_Rp), (const float*)(ws + o_inv),
        (const float*)(ws + o_rm), (const float*)(ws + o_rz), (float*)(ws + o_t));
    // 7) gather + residual
    k_final<<<G(Bn * Cc * HW), TH, 0, stream>>>(input, (const float*)(ws + o_t),
                                                out, Bn * Cc * HW);
}
